// ContentBasedAttention_12180527251775
// MI455X (gfx1250) — matrix operand order repaired and checked by reading, not compiled, not run on hardware
//
#include <hip/hip_runtime.h>
#include <stdint.h>

typedef __attribute__((ext_vector_type(16))) _Float16 v16h;
typedef __attribute__((ext_vector_type(8)))  _Float16 v8h;
typedef __attribute__((ext_vector_type(4)))  _Float16 v4h;
typedef __attribute__((ext_vector_type(8)))  float    v8f;
typedef __attribute__((ext_vector_type(4)))  float    v4f;

// ---------------------------------------------------------------------------
// fp32 -> fp16 elementwise cast (vectorized x4)
// ---------------------------------------------------------------------------
__global__ __launch_bounds__(256) void k_cast_f16(const float* __restrict__ in,
                                                  _Float16* __restrict__ out) {
  long long i = ((long long)blockIdx.x * 256 + threadIdx.x) * 4;
  v4f x = *(const v4f*)(in + i);
  v4h y;
  y[0] = (_Float16)x[0]; y[1] = (_Float16)x[1];
  y[2] = (_Float16)x[2]; y[3] = (_Float16)x[3];
  *(v4h*)(out + i) = y;
}

// ---------------------------------------------------------------------------
// fp32 [R][C] -> fp16 [C][R] tiled transpose-cast (32x32 tiles via LDS)
// ---------------------------------------------------------------------------
__global__ __launch_bounds__(256) void k_transpose_cast(
    const float* __restrict__ in, _Float16* __restrict__ out,
    int R, int C, long long sIn, long long sOut) {
  __shared__ float tile[32][33];
  in  += (long long)blockIdx.z * sIn;
  out += (long long)blockIdx.z * sOut;
  const int c0 = blockIdx.x * 32;
  const int r0 = blockIdx.y * 32;
  const int tx = threadIdx.x & 31;
  const int ty = threadIdx.x >> 5;
#pragma unroll
  for (int p = 0; p < 32; p += 8)
    tile[ty + p][tx] = in[(long long)(r0 + ty + p) * C + c0 + tx];
  __syncthreads();
#pragma unroll
  for (int p = 0; p < 32; p += 8)
    out[(long long)(c0 + ty + p) * R + r0 + tx] = (_Float16)tile[tx][ty + p];
}

// ---------------------------------------------------------------------------
// Row-wise L2 normalize: f32 [rows][D] -> f16 normalized rows (D == 1024)
// ---------------------------------------------------------------------------
__global__ __launch_bounds__(256) void k_rownorm(const float* __restrict__ in,
                                                 _Float16* __restrict__ out,
                                                 int D) {
  const long long row = blockIdx.x;
  const float* p = in + row * (long long)D;
  float ss = 0.f;
  for (int i = threadIdx.x * 4; i < D; i += 1024) {
    v4f x = *(const v4f*)(p + i);
    ss += x[0]*x[0] + x[1]*x[1] + x[2]*x[2] + x[3]*x[3];
  }
  __shared__ float red[256];
  red[threadIdx.x] = ss;
  __syncthreads();
  for (int s = 128; s > 0; s >>= 1) {
    if (threadIdx.x < s) red[threadIdx.x] += red[threadIdx.x + s];
    __syncthreads();
  }
  const float inv = rsqrtf(fmaxf(red[0], 1e-12f));
  _Float16* q = out + row * (long long)D;
  for (int i = threadIdx.x * 4; i < D; i += 1024) {
    v4f x = *(const v4f*)(p + i);
    v4h y;
    y[0] = (_Float16)(x[0]*inv); y[1] = (_Float16)(x[1]*inv);
    y[2] = (_Float16)(x[2]*inv); y[3] = (_Float16)(x[3]*inv);
    *(v4h*)(q + i) = y;
  }
}

// ---------------------------------------------------------------------------
// Rowsum of exp-scores (f16 [rows][W]) -> 1/sum per row (softmax denominator)
// ---------------------------------------------------------------------------
__global__ __launch_bounds__(256) void k_rowsum_inv(const _Float16* __restrict__ P,
                                                    float* __restrict__ inv, int W) {
  const long long row = blockIdx.x;
  const _Float16* p = P + row * (long long)W;
  float s = 0.f;
  for (int i = threadIdx.x * 4; i < W; i += 1024) {
    v4h x = *(const v4h*)(p + i);
    s += (float)x[0] + (float)x[1] + (float)x[2] + (float)x[3];
  }
  __shared__ float red[256];
  red[threadIdx.x] = s;
  __syncthreads();
  for (int st = 128; st > 0; st >>= 1) {
    if (threadIdx.x < st) red[threadIdx.x] += red[threadIdx.x + st];
    __syncthreads();
  }
  if (threadIdx.x == 0) inv[row] = 1.0f / red[0];
}

// ---------------------------------------------------------------------------
// Async global->LDS helper (CDNA5, tracked by ASYNCcnt).
// LDS byte address = low 32 bits of the generic pointer (ISA 10.2).
// ---------------------------------------------------------------------------
__device__ __forceinline__ void async_b128(unsigned lds_addr, const void* gaddr) {
  asm volatile("global_load_async_to_lds_b128 %0, %1, off"
               :: "v"(lds_addr), "v"((unsigned long long)(uintptr_t)gaddr)
               : "memory");
}

// ---------------------------------------------------------------------------
// WMMA GEMM: C[M,N] = A[M,K] @ Bt[N,K]^T, f16 in / f32 accumulate.
// Block tile 128(M) x 256(N), 256 threads = 8 waves as 2(M) x 4(N); each wave
// owns a 64x64 sub-tile = 4x4 WMMA fragments (16 v_wmma per 32-K chunk against
// 16 ds_load_b128 -> 0.5 KB LDS traffic per WMMA).
// Double-buffered LDS fed by global_load_async_to_lds_b128: 2 chunks in
// flight; async loads complete in order, so s_wait_asynccnt 4 releases the
// current chunk while the next keeps streaming.
// MODE 0: store f32 C    MODE 1: store f16 exp(C)    MODE 2: f32 C * invRow
// ---------------------------------------------------------------------------
template <int MODE>
__global__ __launch_bounds__(256) void k_gemm(
    const _Float16* __restrict__ A, const _Float16* __restrict__ Bt,
    void* __restrict__ Cout, const float* __restrict__ invRow,
    int M, int N, int K,
    long long sA, long long sB, long long sC, long long sInv) {
  constexpr int TK = 32;
  constexpr int LS = 40;  // LDS row stride in halfs: 80B, 16B aligned, bank-skewed
  __shared__ _Float16 As[2][128 * LS];
  __shared__ _Float16 Bs[2][256 * LS];

  const int zb = blockIdx.z;
  A  += (long long)zb * sA;
  Bt += (long long)zb * sB;
  const int m0 = blockIdx.y * 128;
  const int n0 = blockIdx.x * 256;
  const int tid  = threadIdx.x;
  const int lane = tid & 31;
  const int wid  = tid >> 5;
  const int wm   = (wid & 1) * 64;   // wave row base in tile
  const int wn   = (wid >> 1) * 64;  // wave col base in tile
  const int half = lane >> 4;        // lane half per ISA fragment layout
  const int l15  = lane & 15;

  v8f acc[4][4];
#pragma unroll
  for (int i = 0; i < 4; ++i)
#pragma unroll
    for (int j = 0; j < 4; ++j) {
      v8f z = {0.f, 0.f, 0.f, 0.f, 0.f, 0.f, 0.f, 0.f};
      acc[i][j] = z;
    }

  // Cooperative tile fill (per thread per chunk: 2x b128 of A, 2x b128 of Bt).
  // A tile: 128 rows x 32 halfs -> thread owns half a row.
  // B tile: 256 rows x 32 halfs -> thread owns one full row.
  const int ldRow = tid >> 1;
  const int ldSeg = (tid & 1) * 16;
  const _Float16* gA = A  + (long long)(m0 + ldRow) * K + ldSeg;
  const _Float16* gB = Bt + (long long)(n0 + tid) * K;
  const unsigned lAo0 = (unsigned)(uintptr_t)(&As[0][0] + ldRow * LS + ldSeg);
  const unsigned lBo0 = (unsigned)(uintptr_t)(&Bs[0][0] + tid * LS);
  const unsigned strideA = (unsigned)(128 * LS * sizeof(_Float16));
  const unsigned strideB = (unsigned)(256 * LS * sizeof(_Float16));

  auto issue = [&](int buf, int kt) {
    const unsigned la = lAo0 + (unsigned)buf * strideA;
    const unsigned lb = lBo0 + (unsigned)buf * strideB;
    async_b128(la,      gA + kt);
    async_b128(la + 16, gA + kt + 8);
    async_b128(lb,      gB + kt);
    async_b128(lb + 16, gB + kt + 8);
  };

  const int nk = K / TK;
  issue(0, 0);
  if (nk > 1) issue(1, TK);

  for (int c = 0; c < nk; ++c) {
    const int cur = c & 1;
    // Release chunk c: with chunk c+1 still in flight, 4 of the 8 outstanding
    // async ops may remain; on the last chunk drain fully.
    if (c + 1 < nk) asm volatile("s_wait_asynccnt 0x4" ::: "memory");
    else            asm volatile("s_wait_asynccnt 0x0" ::: "memory");
    __syncthreads();

    // A fragments (16x32): lane half h holds K = {8h..8h+7} U {16+8h..16+8h+7}
    v16h af[4];
#pragma unroll
    for (int i = 0; i < 4; ++i) {
      const _Float16* p = &As[cur][0] + (wm + i * 16 + l15) * LS + half * 8;
      v8h lo = *(const v8h*)p;
      v8h hi = *(const v8h*)(p + 16);
      af[i] = __builtin_shufflevector(lo, hi, 0,1,2,3,4,5,6,7,8,9,10,11,12,13,14,15);
    }
    // B fragments (32x16): lane half h holds K = {8h..8h+7} U {16+8h..16+8h+7} of column n
    v16h bf[4];
#pragma unroll
    for (int j = 0; j < 4; ++j) {
      const _Float16* p = &Bs[cur][0] + (wn + j * 16 + l15) * LS + half * 8;
      v8h lo = *(const v8h*)p;
      v8h hi = *(const v8h*)(p + 16);
      bf[j] = __builtin_shufflevector(lo, hi, 0,1,2,3,4,5,6,7,8,9,10,11,12,13,14,15);
    }
#pragma unroll
    for (int i = 0; i < 4; ++i)
#pragma unroll
      for (int j = 0; j < 4; ++j)
        acc[i][j] = __builtin_amdgcn_wmma_f32_16x16x32_f16(
            false, af[i], false, bf[j], (short)0, acc[i][j], false, false);

    __syncthreads();  // everyone done reading buf[cur] before it is refilled
    if (c + 2 < nk) issue(cur, (c + 2) * TK);
  }

  // Epilogue. C/D layout: VGPR r -> row r + 8*half, lane -> col l15.
#pragma unroll
  for (int i = 0; i < 4; ++i) {
#pragma unroll
    for (int j = 0; j < 4; ++j) {
      const int col = n0 + wn + j * 16 + l15;
#pragma unroll
      for (int r = 0; r < 8; ++r) {
        const int row = m0 + wm + i * 16 + half * 8 + r;
        const long long idx = (long long)zb * sC + (long long)row * N + col;
        const float v = acc[i][j][r];
        if constexpr (MODE == 0) {
          ((float*)Cout)[idx] = v;
        } else if constexpr (MODE == 1) {
          ((_Float16*)Cout)[idx] = (_Float16)__expf(v);
        } else {
          ((float*)Cout)[idx] = v * invRow[(long long)zb * sInv + row];
        }
      }
    }
  }
}

// ---------------------------------------------------------------------------
// Host orchestration
// ---------------------------------------------------------------------------
extern "C" void kernel_launch(void* const* d_in, const int* in_sizes, int n_in,
                              void* d_out, int out_size, void* d_ws, size_t ws_size,
                              hipStream_t stream) {
  (void)in_sizes; (void)n_in; (void)out_size; (void)ws_size;

  const long long Bn = 4, S = 2048, D = 1024;
  const long long nBSD = Bn * S * D;  // 8,388,608

  const float* x1 = (const float*)d_in[0];
  const float* x2 = (const float*)d_in[1];
  const float* WQ = (const float*)d_in[2];
  const float* WK = (const float*)d_in[3];
  const float* WV = (const float*)d_in[4];
  float* out = (float*)d_out;

  char* ws = (char*)d_ws;
  size_t off = 0;
  auto take = [&](size_t bytes) {
    size_t o = off;
    off = (off + bytes + 255) & ~(size_t)255;
    return o;
  };
  _Float16* x1h = (_Float16*)(ws + take(nBSD * 2));
  _Float16* x2h = (_Float16*)(ws + take(nBSD * 2));
  _Float16* wqT = (_Float16*)(ws + take(D * D * 2));
  _Float16* wkT = (_Float16*)(ws + take(D * D * 2));
  _Float16* wvT = (_Float16*)(ws + take(D * D * 2));
  float*    scr = (float*)   (ws + take(nBSD * 4));      // q/k/v f32 staging (reused)
  _Float16* qnh = (_Float16*)(ws + take(nBSD * 2));
  _Float16* knh = (_Float16*)(ws + take(nBSD * 2));
  _Float16* vTh = (_Float16*)(ws + take(nBSD * 2));      // v^T per batch: [D][S]
  _Float16* Ph  = (_Float16*)(ws + take(Bn * S * S * 2)); // exp(scores) f16
  float*    invR= (float*)   (ws + take(Bn * S * 4));

  // 1) casts + weight transposes
  k_cast_f16<<<(int)(nBSD / 1024), 256, 0, stream>>>(x1, x1h);
  k_cast_f16<<<(int)(nBSD / 1024), 256, 0, stream>>>(x2, x2h);
  dim3 tgw((unsigned)(D / 32), (unsigned)(D / 32), 1);
  k_transpose_cast<<<tgw, 256, 0, stream>>>(WQ, wqT, (int)D, (int)D, 0, 0);
  k_transpose_cast<<<tgw, 256, 0, stream>>>(WK, wkT, (int)D, (int)D, 0, 0);
  k_transpose_cast<<<tgw, 256, 0, stream>>>(WV, wvT, (int)D, (int)D, 0, 0);

  // 2) projections + normalize / transpose
  dim3 g1((unsigned)(D / 256), (unsigned)((Bn * S) / 128), 1);
  k_gemm<0><<<g1, 256, 0, stream>>>(x1h, wqT, scr, nullptr,
                                    (int)(Bn * S), (int)D, (int)D, 0, 0, 0, 0);
  k_rownorm<<<(int)(Bn * S), 256, 0, stream>>>(scr, qnh, (int)D);

  k_gemm<0><<<g1, 256, 0, stream>>>(x2h, wkT, scr, nullptr,
                                    (int)(Bn * S), (int)D, (int)D, 0, 0, 0, 0);
  k_rownorm<<<(int)(Bn * S), 256, 0, stream>>>(scr, knh, (int)D);

  k_gemm<0><<<g1, 256, 0, stream>>>(x2h, wvT, scr, nullptr,
                                    (int)(Bn * S), (int)D, (int)D, 0, 0, 0, 0);
  dim3 tgv((unsigned)(D / 32), (unsigned)(S / 32), (unsigned)Bn);
  k_transpose_cast<<<tgv, 256, 0, stream>>>(scr, vTh, (int)S, (int)D, S * D, S * D);

  // 3) cosine scores -> exp(z) f16 (z in [-1,1], no max-subtraction needed)
  dim3 g2((unsigned)(S / 256), (unsigned)(S / 128), (unsigned)Bn);
  k_gemm<1><<<g2, 256, 0, stream>>>(qnh, knh, Ph, nullptr,
                                    (int)S, (int)S, (int)D, S * D, S * D, S * S, 0);

  // 4) softmax denominator
  k_rowsum_inv<<<(int)(Bn * S), 256, 0, stream>>>(Ph, invR, (int)S);

  // 5) y = (P @ V) / rowsum
  dim3 g3((unsigned)(D / 256), (unsigned)(S / 128), (unsigned)Bn);
  k_gemm<2><<<g3, 256, 0, stream>>>(Ph, vTh, out, invR,
                                    (int)S, (int)D, (int)S, S * S, D * S, S * D, S);
}
